// MiniVITS_1623497638601
// MI455X (gfx1250) — matrix operand order repaired and checked by reading, not compiled, not run on hardware
//
#include <hip/hip_runtime.h>

// ---------------- model dimensions ----------------
#define Bb   32
#define Tt   256
#define Mm   2048
#define MELc 80
#define Hh   192
#define Ll   128
#define CHS  192   // channel stride of all bf16 activation buffers
#define HALO 2     // zero halo positions on each end (max conv pad = 2)

typedef __attribute__((ext_vector_type(16))) __bf16 v16bf;
typedef __attribute__((ext_vector_type(8)))  __bf16 v8bf;
typedef __attribute__((ext_vector_type(8)))  float  v8f;

union ABu { v16bf v; v8bf h[2]; };

// ---------------- utility kernels ----------------
__global__ void k_zero16(ulonglong2* p, size_t n16) {
  size_t i = (size_t)blockIdx.x * blockDim.x + threadIdx.x;
  size_t st = (size_t)gridDim.x * blockDim.x;
  ulonglong2 z; z.x = 0; z.y = 0;
  for (; i < n16; i += st) p[i] = z;
}

// weights (Cout, Cin, K) f32  ->  (K, Cout_p, Cin_p) bf16, zero padded
__global__ void k_convert_w(const float* __restrict__ w, __bf16* __restrict__ o,
                            int Cout, int Cin, int K, int Coutp, int Cinp) {
  long n = (long)K * Coutp * Cinp;
  long i = (long)blockIdx.x * blockDim.x + threadIdx.x;
  long st = (long)gridDim.x * blockDim.x;
  for (; i < n; i += st) {
    int kk = (int)(i / ((long)Coutp * Cinp));
    int r  = (int)(i % ((long)Coutp * Cinp));
    int co = r / Cinp, ci = r % Cinp;
    float v = (co < Cout && ci < Cin) ? w[((long)co * Cin + ci) * K + kk] : 0.0f;
    o[i] = (__bf16)v;
  }
}

// token embedding -> bf16 activation buffer (b, pos, ch), halo layout
__global__ void k_embed(const int* __restrict__ tok, const float* __restrict__ emb,
                        __bf16* __restrict__ out) {
  long n = (long)Bb * Tt * Hh;
  long i = (long)blockIdx.x * blockDim.x + threadIdx.x;
  long st = (long)gridDim.x * blockDim.x;
  for (; i < n; i += st) {
    int h = (int)(i % Hh);
    int t = (int)((i / Hh) % Tt);
    int b = (int)(i / ((long)Hh * Tt));
    float v = emb[(long)tok[b * Tt + t] * Hh + h];
    out[((size_t)b * (Tt + 2 * HALO) + t + HALO) * CHS + h] = (__bf16)v;
  }
}

// mels (b, m, 80) f32 -> bf16 act buffer, channels padded to 96 with zeros
__global__ void k_mels(const float* __restrict__ mels, __bf16* __restrict__ out) {
  long n = (long)Bb * Mm * 96;
  long i = (long)blockIdx.x * blockDim.x + threadIdx.x;
  long st = (long)gridDim.x * blockDim.x;
  for (; i < n; i += st) {
    int c = (int)(i % 96);
    int m = (int)((i / 96) % Mm);
    int b = (int)(i / (96L * Mm));
    float v = (c < MELc) ? mels[((size_t)b * Mm + m) * MELc + c] : 0.0f;
    out[((size_t)b * (Mm + 2 * HALO) + m + HALO) * CHS + c] = (__bf16)v;
  }
}

// ============================================================================
// Cooperative WMMA conv1d: block = 384 threads = 12 waves = all 12 co-tiles of
// Cout=192.  The Cinp x 68-position input slab is staged into LDS once per
// block with gfx1250 async global->LDS copies (ASYNCcnt), then every wave
// feeds its v_wmma_f32_16x16x32_bf16 B-operands from LDS (12x global reuse).
// Output: SiLU + bf16 into halo'd activation buffer.
// ============================================================================
__global__ __launch_bounds__(384) void k_conv_lds(
    const __bf16* __restrict__ act, const __bf16* __restrict__ wgt,
    __bf16* __restrict__ outb, const float* __restrict__ bias,
    int Len, int Cinp, int ks, int pad) {
  __shared__ char smem[68 * CHS * 2];   // max slab: 68 pos x 192 ch bf16
  const int npb  = Len >> 6;
  const int bz   = blockIdx.x / npb;
  const int pblk = (blockIdx.x % npb) * 64;
  const int tid  = threadIdx.x;
  const int lane = tid & 31;
  const int col  = lane & 15;          // WMMA A-row / B-col / D-col index
  const int kh   = lane >> 4;          // K-half selector
  const int cob  = (tid >> 5) * 16;    // wave id -> out-channel tile

  // ---- async stage: slab[pos_local 0..67][ch 0..Cinp) <- global (halo coords) ----
  const int cpp    = Cinp >> 3;        // 16-byte chunks per position
  const int chunks = 68 * cpp;
  const char* actbytes = (const char*)(act + (size_t)bz * (Len + 2 * HALO) * CHS);
  const uint32_t lbase = (uint32_t)(uintptr_t)(void*)smem;  // LDS aperture low bits = LDS offset
  for (int c = tid; c < chunks; c += 384) {
    int p = c / cpp;
    int inner = (c - p * cpp) * 16;
    uint64_t ga = (uint64_t)(uintptr_t)(actbytes + (size_t)(pblk + p) * (CHS * 2) + inner);
    uint32_t la = lbase + (uint32_t)c * 16;
    asm volatile("global_load_async_to_lds_b128 %0, %1, off"
                 :: "v"(la), "v"(ga) : "memory");
  }
#if __has_builtin(__builtin_amdgcn_s_wait_asynccnt)
  __builtin_amdgcn_s_wait_asynccnt(0);
#else
  asm volatile("s_wait_asynccnt 0" ::: "memory");
#endif
  __syncthreads();

  // ---- accumulators seeded with bias ----
  v8f cinit;
#pragma unroll
  for (int v = 0; v < 8; ++v) cinit[v] = bias[cob + 8 * kh + v];
  v8f a0 = cinit, a1 = cinit, a2 = cinit, a3 = cinit;

  const int srow = Cinp * 2;           // LDS bytes per position
  for (int kk = 0; kk < ks; ++kk) {
    // A tile: row = col; elements 0..15 -> K = kh*16+e
    const __bf16* wk = wgt + ((size_t)kk * 192 + cob) * Cinp +
                       (size_t)col * Cinp + kh * 16;
    const char* bp = smem + (size_t)(col + kk - pad + HALO) * srow + kh * 32;
    for (int ck = 0; ck < Cinp; ck += 32) {
      ABu A;
      A.h[0] = *(const v8bf*)(wk + ck);
      A.h[1] = *(const v8bf*)(wk + ck + 8);
      const char* bpc = bp + ck * 2;
      v16bf B0 = *(const v16bf*)(bpc);
      v16bf B1 = *(const v16bf*)(bpc + 16 * srow);
      v16bf B2 = *(const v16bf*)(bpc + 32 * srow);
      v16bf B3 = *(const v16bf*)(bpc + 48 * srow);
      a0 = __builtin_amdgcn_wmma_f32_16x16x32_bf16(false, A.v, false, B0, (short)0, a0, false, false);
      a1 = __builtin_amdgcn_wmma_f32_16x16x32_bf16(false, A.v, false, B1, (short)0, a1, false, false);
      a2 = __builtin_amdgcn_wmma_f32_16x16x32_bf16(false, A.v, false, B2, (short)0, a2, false, false);
      a3 = __builtin_amdgcn_wmma_f32_16x16x32_bf16(false, A.v, false, B3, (short)0, a3, false, false);
    }
  }

  v8f accs[4] = {a0, a1, a2, a3};
  __bf16* ob = outb + ((size_t)bz * (Len + 2 * HALO) + pblk + col + HALO) * CHS +
               cob + 8 * kh;
#pragma unroll
  for (int i = 0; i < 4; ++i) {
    v8bf o;
#pragma unroll
    for (int v = 0; v < 8; ++v) {
      float x = accs[i][v];
      x = x / (1.0f + __expf(-x));     // SiLU
      o[v] = (__bf16)x;
    }
    *(v8bf*)(ob + (size_t)i * 16 * CHS) = o;
  }
}

// ---------------- 1x1 projection conv, f32 output (pm/pls/qm/qls/dp3/dec3) ----
__global__ __launch_bounds__(256) void k_conv1x1(
    const __bf16* __restrict__ act, const __bf16* __restrict__ wgt,
    float* __restrict__ outf, const float* __restrict__ bias,
    int Len, int Cinp, int Coutp, int CoutReal, int ostride, long total_waves) {
  long wave = ((long)blockIdx.x * blockDim.x + threadIdx.x) >> 5;
  if (wave >= total_waves) return;   // whole-wave uniform; EXEC stays all-ones
  const int lane = threadIdx.x & 31;
  const int col  = lane & 15;
  const int kh   = lane >> 4;
  const int npb  = Len >> 6;
  const int nco  = Coutp >> 4;
  const int bz   = (int)(wave / ((long)nco * npb));
  const int r    = (int)(wave % ((long)nco * npb));
  const int cob  = (r / npb) * 16;
  const int pblk = (r % npb) * 64;

  v8f cinit;
#pragma unroll
  for (int v = 0; v < 8; ++v) {
    int co = cob + 8 * kh + v;
    cinit[v] = (co < CoutReal) ? bias[co] : 0.0f;
  }
  v8f a0 = cinit, a1 = cinit, a2 = cinit, a3 = cinit;

  const __bf16* actb = act + (size_t)bz * (Len + 2 * HALO) * CHS;
  const __bf16* wk = wgt + (size_t)cob * Cinp + (size_t)col * Cinp + kh * 16;
  const __bf16* bp0 = actb + (size_t)(pblk + col + HALO) * CHS + kh * 16;
  for (int ck = 0; ck < Cinp; ck += 32) {
    ABu A;
    A.h[0] = *(const v8bf*)(wk + ck);
    A.h[1] = *(const v8bf*)(wk + ck + 8);
    v16bf B0 = *(const v16bf*)(bp0 + ck);
    v16bf B1 = *(const v16bf*)(bp0 + ck + 16 * CHS);
    v16bf B2 = *(const v16bf*)(bp0 + ck + 32 * CHS);
    v16bf B3 = *(const v16bf*)(bp0 + ck + 48 * CHS);
    a0 = __builtin_amdgcn_wmma_f32_16x16x32_bf16(false, A.v, false, B0, (short)0, a0, false, false);
    a1 = __builtin_amdgcn_wmma_f32_16x16x32_bf16(false, A.v, false, B1, (short)0, a1, false, false);
    a2 = __builtin_amdgcn_wmma_f32_16x16x32_bf16(false, A.v, false, B2, (short)0, a2, false, false);
    a3 = __builtin_amdgcn_wmma_f32_16x16x32_bf16(false, A.v, false, B3, (short)0, a3, false, false);
  }

  v8f accs[4] = {a0, a1, a2, a3};
  float* of = outf + ((size_t)bz * Len + pblk + col) * ostride + cob + 8 * kh;
#pragma unroll
  for (int i = 0; i < 4; ++i) {
    *(float4*)(of + (size_t)i * 16 * ostride) =
        make_float4(accs[i][0], accs[i][1], accs[i][2], accs[i][3]);
    *(float4*)(of + (size_t)i * 16 * ostride + 4) =
        make_float4(accs[i][4], accs[i][5], accs[i][6], accs[i][7]);
  }
}

// ---------------- alignment path (hard monotonic gather) ----------------
__global__ void k_alignment(const int* __restrict__ tlen, const int* __restrict__ mlen,
                            float* __restrict__ dur, int* __restrict__ tofm) {
  int b = blockIdx.x;
  float tl = (float)tlen[b], ml = (float)mlen[b];
  int tli = tlen[b], mli = mlen[b];
  for (int t = threadIdx.x; t < Tt; t += blockDim.x) {
    float s = roundf((float)t * ml / tl);
    float e = roundf((float)(t + 1) * ml / tl);
    if (e <= s) e = fminf(s + 1.0f, ml);
    dur[b * Tt + t] = (t < tli) ? (e - s) : 0.0f;
  }
  for (int m = threadIdx.x; m < Mm; m += blockDim.x) {
    if (m < mli) {
      int g = (int)(((float)m) * tl / ml);
      if (g > tli - 1) g = tli - 1;
      if (g < 0) g = 0;
      while (g + 1 < tli && (float)m >= roundf((float)(g + 1) * ml / tl)) ++g;
      while (g > 0 && (float)m < roundf((float)g * ml / tl)) --g;
      tofm[b * Mm + m] = g;
    }
  }
}

// latents = qm + noise * exp(qls)  -> bf16 act buffer (decoder input, 128 ch)
__global__ void k_latents(const float* __restrict__ qm, const float* __restrict__ qls,
                          const float* __restrict__ noise, __bf16* __restrict__ out) {
  long n = (long)Bb * Mm * Ll;
  long i = (long)blockIdx.x * blockDim.x + threadIdx.x;
  long st = (long)gridDim.x * blockDim.x;
  for (; i < n; i += st) {
    int l = (int)(i % Ll);
    int m = (int)((i / Ll) % Mm);
    int b = (int)(i / ((long)Ll * Mm));
    size_t qi = ((size_t)b * Mm + m) * Ll + l;
    float v = qm[qi] + noise[(size_t)b * Ll * Mm + (size_t)l * Mm + m] * expf(qls[qi]);
    out[((size_t)b * (Mm + 2 * HALO) + m + HALO) * CHS + l] = (__bf16)v;
  }
}

// ---------------- reductions ----------------
__device__ __forceinline__ float block_reduce(float v) {
  __shared__ float s[256];
  s[threadIdx.x] = v;
  __syncthreads();
  for (int o = 128; o > 0; o >>= 1) {
    if (threadIdx.x < (unsigned)o) s[threadIdx.x] += s[threadIdx.x + o];
    __syncthreads();
  }
  return s[0];
}

__global__ void k_mel_loss(const float* __restrict__ pred /* d_out+4 */,
                           const float* __restrict__ mels,
                           const int* __restrict__ mlen, float* __restrict__ acc) {
  long n = (long)Bb * Mm * MELc;
  long idx = (long)blockIdx.x * blockDim.x + threadIdx.x;
  float v = 0.0f;
  if (idx < n) {
    int m = (int)((idx / MELc) % Mm);
    int b = (int)(idx / ((long)MELc * Mm));
    if (m < mlen[b]) v = fabsf(pred[idx] - mels[idx]);
  }
  float s = block_reduce(v);
  if (threadIdx.x == 0) atomicAdd(&acc[0], s);
}

__global__ void k_dur_loss(const float* __restrict__ dpf, const float* __restrict__ dur,
                           const int* __restrict__ tlen, float* __restrict__ acc) {
  long idx = (long)blockIdx.x * blockDim.x + threadIdx.x;
  float v = 0.0f;
  if (idx < (long)Bb * Tt) {
    int t = (int)(idx % Tt);
    int b = (int)(idx / Tt);
    if (t < tlen[b]) {
      float pred = dpf[idx * 16];           // channel 0 of padded dp3 output
      float tgt  = logf(dur[idx] + 1.0f);
      float d = pred - tgt;
      v = d * d;
    }
  }
  float s = block_reduce(v);
  if (threadIdx.x == 0) atomicAdd(&acc[1], s);
}

__global__ void k_kl(const float* __restrict__ pm, const float* __restrict__ pls,
                     const float* __restrict__ qm, const float* __restrict__ qls,
                     const int* __restrict__ tofm, const int* __restrict__ mlen,
                     float* __restrict__ acc) {
  long n = (long)Bb * Mm * Ll;
  long idx = (long)blockIdx.x * blockDim.x + threadIdx.x;
  float v = 0.0f;
  if (idx < n) {
    int l = (int)(idx % Ll);
    int m = (int)((idx / Ll) % Mm);
    int b = (int)(idx / ((long)Ll * Mm));
    if (m < mlen[b]) {
      int t = tofm[b * Mm + m];
      size_t pi = ((size_t)b * Tt + t) * Ll + l;
      size_t qi = ((size_t)b * Mm + m) * Ll + l;
      float am = pm[pi], als = pls[pi];
      float pmu = qm[qi], plsv = qls[qi];
      float d = pmu - am;
      v = als - plsv + (expf(2.0f * plsv) + d * d) / (2.0f * expf(2.0f * als)) - 0.5f;
    }
  }
  float s = block_reduce(v);
  if (threadIdx.x == 0) atomicAdd(&acc[2], s);
}

__global__ void k_final(const float* __restrict__ acc, const int* __restrict__ mlen,
                        float* __restrict__ out) {
  if (threadIdx.x == 0 && blockIdx.x == 0) {
    float mel = acc[0] / (float)((long)Bb * Mm * MELc);
    float dl  = acc[1] / (float)(Bb * Tt);
    float ms = 0.0f;
    for (int b = 0; b < Bb; ++b) ms += (float)mlen[b];
    float kl = acc[2] / fmaxf(ms, 1.0f);
    out[0] = mel + 0.1f * dl + 0.01f * kl;
    out[1] = mel;
    out[2] = dl;
    out[3] = kl;
  }
}

// ---------------- host orchestration ----------------
static inline void zero_buf(hipStream_t s, void* p, size_t bytes) {
  size_t n16 = bytes / 16;
  int blocks = (int)((n16 + 255) / 256);
  if (blocks > 32768) blocks = 32768;
  if (blocks < 1) blocks = 1;
  k_zero16<<<blocks, 256, 0, s>>>((ulonglong2*)p, n16);
}

static inline void cvt_w(hipStream_t s, const void* w, void* o,
                         int Cout, int Cin, int K, int Coutp, int Cinp) {
  long n = (long)K * Coutp * Cinp;
  int blocks = (int)((n + 255) / 256);
  k_convert_w<<<blocks, 256, 0, s>>>((const float*)w, (__bf16*)o, Cout, Cin, K, Coutp, Cinp);
}

// bf16+SiLU conv, Cout fixed at 192, cooperative LDS-staged kernel
static inline void conv_bf16(hipStream_t s, const void* in, const void* w, void* out,
                             const float* bias, int Len, int Cinp, int ks, int pad) {
  int blocks = Bb * (Len / 64);
  k_conv_lds<<<blocks, 384, 0, s>>>((const __bf16*)in, (const __bf16*)w, (__bf16*)out,
                                    bias, Len, Cinp, ks, pad);
}

static inline void conv_f32(hipStream_t s, const void* in, const void* w, float* out,
                            const float* bias, int Len, int Cinp, int Coutp, int CoutReal,
                            int ostride) {
  long waves = (long)Bb * (Coutp / 16) * (Len / 64);
  int blocks = (int)((waves + 7) / 8);
  k_conv1x1<<<blocks, 256, 0, s>>>((const __bf16*)in, (const __bf16*)w, out, bias,
                                   Len, Cinp, Coutp, CoutReal, ostride, waves);
}

extern "C" void kernel_launch(void* const* d_in, const int* in_sizes, int n_in,
                              void* d_out, int out_size, void* d_ws, size_t ws_size,
                              hipStream_t stream) {
  (void)in_sizes; (void)n_in; (void)out_size; (void)ws_size;
  // ---- inputs (setup_inputs dict order, params flattened in insertion order) ----
  const int*   tokens = (const int*)d_in[0];
  const int*   tlen   = (const int*)d_in[1];
  const float* mels   = (const float*)d_in[2];
  const int*   mlen   = (const int*)d_in[3];
  const float* noise  = (const float*)d_in[4];
  const float* emb    = (const float*)d_in[5];
  // te_w 6,7,8  te_b 9,10,11  pm_w 12 pm_b 13  pls_w 14 pls_b 15
  // pe_w 16,17,18  pe_b 19,20,21  qm_w 22 qm_b 23  qls_w 24 qls_b 25
  // dp_w 26,27,28  dp_b 29,30,31  dec_w 32,33,34  dec_b 35,36,37
  float* out = (float*)d_out;

  // ---- workspace layout ----
  size_t off = 0;
  auto alloc = [&](size_t bytes) -> char* {
    off = (off + 255) & ~(size_t)255;
    char* p = (char*)d_ws + off;
    off += bytes;
    return p;
  };
  char* wte1  = alloc((size_t)5 * 192 * 192 * 2);
  char* wte2  = alloc((size_t)5 * 192 * 192 * 2);
  char* wte3  = alloc((size_t)5 * 192 * 192 * 2);
  char* wpe1  = alloc((size_t)5 * 192 * 96 * 2);
  char* wpe2  = alloc((size_t)5 * 192 * 192 * 2);
  char* wpe3  = alloc((size_t)5 * 192 * 192 * 2);
  char* wdec1 = alloc((size_t)5 * 192 * 128 * 2);
  char* wdec2 = alloc((size_t)5 * 192 * 192 * 2);
  char* wdec3 = alloc((size_t)1 * 80 * 192 * 2);
  char* wpm   = alloc((size_t)1 * 128 * 192 * 2);
  char* wpls  = alloc((size_t)1 * 128 * 192 * 2);
  char* wqm   = alloc((size_t)1 * 128 * 192 * 2);
  char* wqls  = alloc((size_t)1 * 128 * 192 * 2);
  char* wdp1  = alloc((size_t)3 * 192 * 192 * 2);
  char* wdp2  = alloc((size_t)3 * 192 * 192 * 2);
  char* wdp3  = alloc((size_t)1 * 16 * 192 * 2);

  size_t actT_bytes = (size_t)Bb * (Tt + 2 * HALO) * CHS * 2;
  size_t actM_bytes = (size_t)Bb * (Mm + 2 * HALO) * CHS * 2;
  char* actTA = alloc(actT_bytes);
  char* actTB = alloc(actT_bytes);
  char* actMA = alloc(actM_bytes);
  char* actMB = alloc(actM_bytes);

  float* pmf  = (float*)alloc((size_t)Bb * Tt * Ll * 4);
  float* plsf = (float*)alloc((size_t)Bb * Tt * Ll * 4);
  float* qmf  = (float*)alloc((size_t)Bb * Mm * Ll * 4);
  float* qlsf = (float*)alloc((size_t)Bb * Mm * Ll * 4);
  float* dpf  = (float*)alloc((size_t)Bb * Tt * 16 * 4);
  float* durf = (float*)alloc((size_t)Bb * Tt * 4);
  int*   tofm = (int*)alloc((size_t)Bb * Mm * 4);
  float* acc  = (float*)alloc(256);

  // ---- zero halo'd activation buffers + accumulators (every call: deterministic) ----
  zero_buf(stream, actTA, actT_bytes);
  zero_buf(stream, actTB, actT_bytes);
  zero_buf(stream, actMA, actM_bytes);
  zero_buf(stream, actMB, actM_bytes);
  zero_buf(stream, acc, 256);

  // ---- repack weights to (tap, Cout_p, Cin_p) bf16 ----
  cvt_w(stream, d_in[6], wte1, 192, 192, 5, 192, 192);
  cvt_w(stream, d_in[7], wte2, 192, 192, 5, 192, 192);
  cvt_w(stream, d_in[8], wte3, 192, 192, 5, 192, 192);
  cvt_w(stream, d_in[16], wpe1, 192, 80, 5, 192, 96);
  cvt_w(stream, d_in[17], wpe2, 192, 192, 5, 192, 192);
  cvt_w(stream, d_in[18], wpe3, 192, 192, 5, 192, 192);
  cvt_w(stream, d_in[32], wdec1, 192, 128, 5, 192, 128);
  cvt_w(stream, d_in[33], wdec2, 192, 192, 5, 192, 192);
  cvt_w(stream, d_in[34], wdec3, 80, 192, 1, 80, 192);
  cvt_w(stream, d_in[12], wpm, 128, 192, 1, 128, 192);
  cvt_w(stream, d_in[14], wpls, 128, 192, 1, 128, 192);
  cvt_w(stream, d_in[22], wqm, 128, 192, 1, 128, 192);
  cvt_w(stream, d_in[24], wqls, 128, 192, 1, 128, 192);
  cvt_w(stream, d_in[26], wdp1, 192, 192, 3, 192, 192);
  cvt_w(stream, d_in[27], wdp2, 192, 192, 3, 192, 192);
  cvt_w(stream, d_in[28], wdp3, 1, 192, 1, 16, 192);

  // ---- text path ----
  {
    long n = (long)Bb * Tt * Hh;
    k_embed<<<(int)((n + 255) / 256), 256, 0, stream>>>(tokens, emb, (__bf16*)actTA);
  }
  conv_bf16(stream, actTA, wte1, actTB, (const float*)d_in[9], Tt, 192, 5, 2);
  conv_bf16(stream, actTB, wte2, actTA, (const float*)d_in[10], Tt, 192, 5, 2);
  conv_bf16(stream, actTA, wte3, actTB, (const float*)d_in[11], Tt, 192, 5, 2); // hidden in TB
  conv_f32(stream, actTB, wpm, pmf, (const float*)d_in[13], Tt, 192, 128, 128, 128);
  conv_f32(stream, actTB, wpls, plsf, (const float*)d_in[15], Tt, 192, 128, 128, 128);
  conv_bf16(stream, actTB, wdp1, actTA, (const float*)d_in[29], Tt, 192, 3, 1);
  conv_bf16(stream, actTA, wdp2, actTB, (const float*)d_in[30], Tt, 192, 3, 1);
  conv_f32(stream, actTB, wdp3, dpf, (const float*)d_in[31], Tt, 192, 16, 1, 16);

  // ---- posterior path ----
  {
    long n = (long)Bb * Mm * 96;
    k_mels<<<(int)((n + 255) / 256), 256, 0, stream>>>(mels, (__bf16*)actMA);
  }
  conv_bf16(stream, actMA, wpe1, actMB, (const float*)d_in[19], Mm, 96, 5, 2);
  conv_bf16(stream, actMB, wpe2, actMA, (const float*)d_in[20], Mm, 192, 5, 2);
  conv_bf16(stream, actMA, wpe3, actMB, (const float*)d_in[21], Mm, 192, 5, 2);
  conv_f32(stream, actMB, wqm, qmf, (const float*)d_in[23], Mm, 192, 128, 128, 128);
  conv_f32(stream, actMB, wqls, qlsf, (const float*)d_in[25], Mm, 192, 128, 128, 128);

  // ---- alignment + latents ----
  k_alignment<<<Bb, 256, 0, stream>>>(tlen, mlen, durf, tofm);
  {
    long n = (long)Bb * Mm * Ll;
    k_latents<<<(int)((n + 255) / 256), 256, 0, stream>>>(qmf, qlsf, noise, (__bf16*)actMA);
  }

  // ---- decoder ----
  conv_bf16(stream, actMA, wdec1, actMB, (const float*)d_in[35], Mm, 128, 5, 2);
  conv_bf16(stream, actMB, wdec2, actMA, (const float*)d_in[36], Mm, 192, 5, 2);
  conv_f32(stream, actMA, wdec3, out + 4, (const float*)d_in[37], Mm, 192, 80, 80, 80);

  // ---- losses ----
  {
    long n = (long)Bb * Mm * MELc;
    k_mel_loss<<<(int)((n + 255) / 256), 256, 0, stream>>>(out + 4, mels, mlen, acc);
  }
  {
    long n = (long)Bb * Tt;
    k_dur_loss<<<(int)((n + 255) / 256), 256, 0, stream>>>(dpf, durf, tlen, acc);
  }
  {
    long n = (long)Bb * Mm * Ll;
    k_kl<<<(int)((n + 255) / 256), 256, 0, stream>>>(pmf, plsf, qmf, qlsf, tofm, mlen, acc);
  }
  k_final<<<1, 64, 0, stream>>>(acc, mlen, out);
}
